// Model_23038204575811
// MI455X (gfx1250) — matrix operand order repaired and checked by reading, not compiled, not run on hardware
//
#include <hip/hip_runtime.h>
#include <hip/hip_bf16.h>

// ---------------------------------------------------------------------------
// Types for CDNA5 WMMA (wave32, v_wmma_f32_16x16x32_bf16)
// ---------------------------------------------------------------------------
typedef __attribute__((ext_vector_type(16))) __bf16        bf16x16;
typedef __attribute__((ext_vector_type(8)))  float         f32x8;
typedef __attribute__((ext_vector_type(4)))  unsigned int  u32x4;

union Frag16 {
    u32x4   u[2];   // 2 x 16B loads
    bf16x16 bf;     // WMMA operand view
};

__device__ __forceinline__ unsigned short f2bf(float f) {
    union { float f; unsigned int u; } c; c.f = f;
    unsigned int u = c.u;
    unsigned int r = u + 0x7FFFu + ((u >> 16) & 1u);      // round-to-nearest-even
    if ((u & 0x7F800000u) == 0x7F800000u) r = u;          // inf/nan passthrough
    return (unsigned short)(r >> 16);
}

#define NEG_INF (-3.402823e38f)

// ---------------------------------------------------------------------------
// GEMM_NT: C[M,N](f32) = A[M,K](bf16,row-major) * B[N,K](bf16,row-major)^T
// Each wave computes a 16(M) x 64(N) strip: one A fragment feeds 4 WMMAs with
// independent accumulators (A traffic /4, back-to-back v_wmma issue).
// 8 waves per block. K multiple of 32, M mult of 16, N mult of 64.
// A fragment (16x32 bf16): lane = kh*16+m ; elems = A[m][k0+kh*16+0..15]
// B fragment (32x16 bf16): lane = kh*16+n ; elems = B[n][k0+kh*16+0..15]
// D (16x16 f32): D[kh*8+j][lane%16] = acc[j]
// ---------------------------------------------------------------------------
__global__ __launch_bounds__(256) void gemm_nt_bf16(
    const unsigned short* __restrict__ A,
    const unsigned short* __restrict__ B,
    float* __restrict__ C,
    int M, int N, int K)
{
    const int wave = threadIdx.x >> 5;
    const int lane = threadIdx.x & 31;
    const int stripsN = N >> 6;                     // 64-wide N strips
    const int stripsTotal = (M >> 4) * stripsN;
    const int strip = blockIdx.x * 8 + wave;
    if (strip >= stripsTotal) return;
    const int tM = (strip / stripsN) << 4;
    const int tN = (strip % stripsN) << 6;

    const int r  = lane & 15;   // m for A, n for B/C
    const int kh = lane >> 4;   // half selector

    const unsigned short* arow = A + (size_t)(tM + r) * K + kh * 16;
    const unsigned short* b0   = B + (size_t)(tN +  0 + r) * K + kh * 16;
    const unsigned short* b1   = B + (size_t)(tN + 16 + r) * K + kh * 16;
    const unsigned short* b2   = B + (size_t)(tN + 32 + r) * K + kh * 16;
    const unsigned short* b3   = B + (size_t)(tN + 48 + r) * K + kh * 16;

    f32x8 acc0 = {}, acc1 = {}, acc2 = {}, acc3 = {};
    for (int k0 = 0; k0 < K; k0 += 32) {
        Frag16 a, f0, f1, f2, f3;
        a.u[0]  = *(const u32x4*)(arow + k0);
        a.u[1]  = *(const u32x4*)(arow + k0 + 8);
        f0.u[0] = *(const u32x4*)(b0 + k0);
        f0.u[1] = *(const u32x4*)(b0 + k0 + 8);
        f1.u[0] = *(const u32x4*)(b1 + k0);
        f1.u[1] = *(const u32x4*)(b1 + k0 + 8);
        f2.u[0] = *(const u32x4*)(b2 + k0);
        f2.u[1] = *(const u32x4*)(b2 + k0 + 8);
        f3.u[0] = *(const u32x4*)(b3 + k0);
        f3.u[1] = *(const u32x4*)(b3 + k0 + 8);
        __builtin_prefetch(arow + k0 + 64, 0, 3);    // global_prefetch_b8
        __builtin_prefetch(b0 + k0 + 64, 0, 3);
        __builtin_prefetch(b2 + k0 + 64, 0, 3);
        acc0 = __builtin_amdgcn_wmma_f32_16x16x32_bf16(
            false, a.bf, false, f0.bf, (short)0, acc0, false, false);
        acc1 = __builtin_amdgcn_wmma_f32_16x16x32_bf16(
            false, a.bf, false, f1.bf, (short)0, acc1, false, false);
        acc2 = __builtin_amdgcn_wmma_f32_16x16x32_bf16(
            false, a.bf, false, f2.bf, (short)0, acc2, false, false);
        acc3 = __builtin_amdgcn_wmma_f32_16x16x32_bf16(
            false, a.bf, false, f3.bf, (short)0, acc3, false, false);
    }

    float* crow = C + (size_t)(tM + kh * 8) * N + tN + r;
#pragma unroll
    for (int j = 0; j < 8; ++j) {
        crow[(size_t)j * N +  0] = acc0[j];
        crow[(size_t)j * N + 16] = acc1[j];
        crow[(size_t)j * N + 32] = acc2[j];
        crow[(size_t)j * N + 48] = acc3[j];
    }
}

// ---------------------------------------------------------------------------
// CNN backbone: two-pass conv3x3(SAME) + batch BN + ReLU + maxpool2
// ---------------------------------------------------------------------------
__device__ __forceinline__ float conv3x3_at(
    const float* __restrict__ in, const float* __restrict__ wco,
    int n, int Cin, int H, int W, int y, int x)
{
    float acc = 0.f;
    const size_t plane = (size_t)H * W;
    const float* base = in + (size_t)n * Cin * plane;
    for (int ci = 0; ci < Cin; ++ci) {
        const float* ip = base + (size_t)ci * plane;
        const float* wp = wco + ci * 9;
#pragma unroll
        for (int ky = 0; ky < 3; ++ky) {
            int yy = y + ky - 1;
            if ((unsigned)yy >= (unsigned)H) continue;
            const float* iprow = ip + (size_t)yy * W;
#pragma unroll
            for (int kx = 0; kx < 3; ++kx) {
                int xx = x + kx - 1;
                if ((unsigned)xx >= (unsigned)W) continue;
                acc = fmaf(iprow[xx], wp[ky * 3 + kx], acc);
            }
        }
    }
    return acc;
}

__global__ void zero_f32(float* p, int n) {
    int i = blockIdx.x * 256 + threadIdx.x;
    if (i < n) p[i] = 0.f;
}

// Pass 1: per-channel sum / sumsq of conv output (Cout=64). block=(n,co) plane.
__global__ __launch_bounds__(256) void conv_stats(
    const float* __restrict__ in, const float* __restrict__ w,
    float* __restrict__ sums, int Cin, int H, int W)
{
    const int n  = blockIdx.x >> 6;
    const int co = blockIdx.x & 63;
    const int HW = H * W;
    const float* wco = w + (size_t)co * Cin * 9;
    float s = 0.f, s2 = 0.f;
    for (int p = threadIdx.x; p < HW; p += 256) {
        int y = p / W, x = p % W;
        float v = conv3x3_at(in, wco, n, Cin, H, W, y, x);
        s += v; s2 += v * v;
    }
    __shared__ float rs[256], rs2[256];
    rs[threadIdx.x] = s; rs2[threadIdx.x] = s2;
    __syncthreads();
    for (int st = 128; st > 0; st >>= 1) {
        if (threadIdx.x < st) {
            rs[threadIdx.x]  += rs[threadIdx.x + st];
            rs2[threadIdx.x] += rs2[threadIdx.x + st];
        }
        __syncthreads();
    }
    if (threadIdx.x == 0) {
        atomicAdd(&sums[co],      rs[0]);
        atomicAdd(&sums[64 + co], rs2[0]);
    }
}

__global__ void finalize_stats(const float* sums, float* stats, float invCount) {
    int c = threadIdx.x;   // 64 threads
    float mu  = sums[c] * invCount;
    float var = sums[64 + c] * invCount - mu * mu;
    stats[c]      = mu;
    stats[64 + c] = rsqrtf(var + 1e-5f);
}

// Pass 2: recompute conv, BN, ReLU, 2x2 maxpool. One thread per pooled output.
__global__ __launch_bounds__(256) void conv_bn_pool(
    const float* __restrict__ in, const float* __restrict__ w,
    const float* __restrict__ stats, const float* __restrict__ g,
    const float* __restrict__ b, float* __restrict__ out,
    int N, int Cin, int H, int W)
{
    const int Ho = H >> 1, Wo = W >> 1;
    const size_t total = (size_t)N * 64 * Ho * Wo;
    size_t idx = (size_t)blockIdx.x * 256 + threadIdx.x;
    if (idx >= total) return;
    int wo = (int)(idx % Wo); size_t t = idx / Wo;
    int ho = (int)(t % Ho);   t /= Ho;
    int co = (int)(t % 64);   int n = (int)(t / 64);

    const float* wco = w + (size_t)co * Cin * 9;
    const float mu = stats[co], rstd = stats[64 + co];
    const float gg = g[co], bb = b[co];
    float m = NEG_INF;
#pragma unroll
    for (int dy = 0; dy < 2; ++dy)
#pragma unroll
        for (int dx = 0; dx < 2; ++dx) {
            float v = conv3x3_at(in, wco, n, Cin, H, W, 2 * ho + dy, 2 * wo + dx);
            v = (v - mu) * rstd * gg + bb;
            v = v > 0.f ? v : 0.f;
            m = v > m ? v : m;
        }
    out[idx] = m;
}

// ---------------------------------------------------------------------------
// Graph construction
// ---------------------------------------------------------------------------
__global__ __launch_bounds__(256) void rownorm_bf16(
    const float* __restrict__ z, unsigned short* __restrict__ znbf, int cols)
{
    const int row = blockIdx.x;
    const float* zr = z + (size_t)row * cols;
    float s = 0.f;
    for (int c = threadIdx.x; c < cols; c += 256) { float v = zr[c]; s += v * v; }
    __shared__ float red[256];
    red[threadIdx.x] = s; __syncthreads();
    for (int st = 128; st > 0; st >>= 1) {
        if (threadIdx.x < st) red[threadIdx.x] += red[threadIdx.x + st];
        __syncthreads();
    }
    const float r = rsqrtf(red[0] + 1e-12f);
    for (int c = threadIdx.x; c < cols; c += 256)
        znbf[(size_t)row * cols + c] = f2bf(zr[c] * r);
}

// top-64 of a 256-wide row by iterative argmax (stable: smaller index wins ties)
__global__ __launch_bounds__(256) void topk64(
    const float* __restrict__ sim, int* __restrict__ nbr)
{
    const int row = blockIdx.x, t = threadIdx.x;
    __shared__ float sv[256], rv[256];
    __shared__ int   ri[256], win;
    float v = sim[(size_t)row * 256 + t];
    if (t == row) v -= 2.0f;                 // sim - 2*eye
    sv[t] = v;
    __syncthreads();
    for (int k = 0; k < 64; ++k) {
        rv[t] = sv[t]; ri[t] = t;
        __syncthreads();
        for (int st = 128; st > 0; st >>= 1) {
            if (t < st) {
                float a = rv[t], b = rv[t + st];
                if (b > a || (b == a && ri[t + st] < ri[t])) {
                    rv[t] = b; ri[t] = ri[t + st];
                }
            }
            __syncthreads();
        }
        if (t == 0) { nbr[row * 64 + k] = ri[0]; win = ri[0]; }
        __syncthreads();
        if (t == win) sv[t] = NEG_INF;
        __syncthreads();
    }
}

// ---------------------------------------------------------------------------
// Fused per-node attention: scores + softmax(K=64) + aggregation. D=1600, H=8.
// ---------------------------------------------------------------------------
__global__ __launch_bounds__(256) void attn_fused(
    const float* __restrict__ q, const float* __restrict__ kk,
    const float* __restrict__ vv, const int* __restrict__ nbr,
    float* __restrict__ m)
{
    __shared__ float qs[1600];
    __shared__ int   nb[64];
    __shared__ float sc[512];        // s[k][h]
    __shared__ float al[512];        // softmax weights
    __shared__ float hmax[8], hsum[8];
    const int n = blockIdx.x, t = threadIdx.x;
    for (int i = t; i < 1600; i += 256) qs[i] = q[(size_t)n * 1600 + i];
    if (t < 64) nb[t] = nbr[n * 64 + t];
    __syncthreads();

    const float scale = 0.07071067811865475f;   // 1/sqrt(200)
    for (int p = t; p < 512; p += 256) {
        const int k = p >> 3, h = p & 7;
        const float* kr = kk + (size_t)nb[k] * 1600 + h * 200;
        const float* qr = qs + h * 200;
        float s = 0.f;
        for (int d = 0; d < 200; ++d) s = fmaf(qr[d], kr[d], s);
        sc[p] = s * scale;
    }
    __syncthreads();
    if (t < 8) {
        float mx = NEG_INF;
        for (int k = 0; k < 64; ++k) mx = fmaxf(mx, sc[k * 8 + t]);
        float sm = 0.f;
        for (int k = 0; k < 64; ++k) sm += __expf(sc[k * 8 + t] - mx);
        hmax[t] = mx; hsum[t] = sm;
    }
    __syncthreads();
    for (int p = t; p < 512; p += 256) {
        const int h = p & 7;
        al[p] = __expf(sc[p] - hmax[h]) / hsum[h];
    }
    __syncthreads();
    for (int i = t; i < 1600; i += 256) {
        const int h = i / 200;
        float acc = 0.f;
        for (int k = 0; k < 64; ++k)
            acc = fmaf(al[k * 8 + h], vv[(size_t)nb[k] * 1600 + i], acc);
        m[(size_t)n * 1600 + i] = acc;
    }
}

// ---------------------------------------------------------------------------
// LayerNorm(resid + x [+ bias]) with optional bf16 mirror output
// ---------------------------------------------------------------------------
__global__ __launch_bounds__(256) void ln_residual(
    const float* __restrict__ resid, const float* __restrict__ x,
    const float* __restrict__ bias, const float* __restrict__ g,
    const float* __restrict__ bb, float* __restrict__ out,
    unsigned short* __restrict__ outbf, int cols)
{
    const int row = blockIdx.x, t = threadIdx.x;
    __shared__ float buf[1600];
    __shared__ float red[256];
    float s = 0.f;
    for (int c = t; c < cols; c += 256) {
        float v = resid[(size_t)row * cols + c] + x[(size_t)row * cols + c]
                + (bias ? bias[c] : 0.f);
        buf[c] = v; s += v;
    }
    red[t] = s; __syncthreads();
    for (int st = 128; st > 0; st >>= 1) {
        if (t < st) red[t] += red[t + st];
        __syncthreads();
    }
    const float mu = red[0] / cols;
    __syncthreads();
    s = 0.f;
    for (int c = t; c < cols; c += 256) { float d = buf[c] - mu; s += d * d; }
    red[t] = s; __syncthreads();
    for (int st = 128; st > 0; st >>= 1) {
        if (t < st) red[t] += red[t + st];
        __syncthreads();
    }
    const float rstd = rsqrtf(red[0] / cols + 1e-5f);
    for (int c = t; c < cols; c += 256) {
        float y = (buf[c] - mu) * rstd * g[c] + bb[c];
        out[(size_t)row * cols + c] = y;
        if (outbf) outbf[(size_t)row * cols + c] = f2bf(y);
    }
}

// ---------------------------------------------------------------------------
// Elementwise helpers
// ---------------------------------------------------------------------------
__global__ void to_bf16(const float* __restrict__ x, unsigned short* __restrict__ y, int n) {
    int i = blockIdx.x * 256 + threadIdx.x;
    if (i < n) y[i] = f2bf(x[i]);
}

// W[K][N] f32 -> Wt[N][K] bf16
__global__ void transpose_bf16(const float* __restrict__ W, unsigned short* __restrict__ Wt,
                               int K, int N) {
    size_t i = (size_t)blockIdx.x * 256 + threadIdx.x;
    if (i >= (size_t)K * N) return;
    int k = (int)(i / N), n = (int)(i % N);
    Wt[(size_t)n * K + k] = f2bf(W[i]);
}

__global__ void bias_relu_bf16(float* __restrict__ x, const float* __restrict__ bias,
                               unsigned short* __restrict__ xbf, int cols, int total) {
    int i = blockIdx.x * 256 + threadIdx.x;
    if (i >= total) return;
    float v = x[i] + bias[i % cols];
    v = v > 0.f ? v : 0.f;
    x[i] = v;
    xbf[i] = f2bf(v);
}

// ---------------------------------------------------------------------------
// Orchestration
// ---------------------------------------------------------------------------
static inline void launch_gemm(const unsigned short* A, const unsigned short* B,
                               float* C, int M, int N, int K, hipStream_t s) {
    int strips = (M >> 4) * (N >> 6);
    gemm_nt_bf16<<<(strips + 7) / 8, 256, 0, s>>>(A, B, C, M, N, K);
}

extern "C" void kernel_launch(void* const* d_in, const int* in_sizes, int n_in,
                              void* d_out, int out_size, void* d_ws, size_t ws_size,
                              hipStream_t stream) {
    const int NB = 256, D = 1600, DD = D * D, ND = NB * D;   // ND = 409600

    const float* x       = (const float*)d_in[0];
    const float* conv_w1 = (const float*)d_in[1];
    const float* conv_w  = (const float*)d_in[2];
    const float* bn_g    = (const float*)d_in[3];
    const float* bn_b    = (const float*)d_in[4];
    const float* wq      = (const float*)d_in[5];
    const float* wk      = (const float*)d_in[6];
    const float* wv      = (const float*)d_in[7];
    const float* wo      = (const float*)d_in[8];
    const float* ln1_g   = (const float*)d_in[9];
    const float* ln1_b   = (const float*)d_in[10];
    const float* mlp_w1  = (const float*)d_in[11];
    const float* mlp_b1  = (const float*)d_in[12];
    const float* mlp_w2  = (const float*)d_in[13];
    const float* mlp_b2  = (const float*)d_in[14];
    const float* ln2_g   = (const float*)d_in[15];
    const float* ln2_b   = (const float*)d_in[16];
    float* out = (float*)d_out;

    char* p = (char*)d_ws;
    auto carve = [&](size_t bytes) -> void* {
        void* r = (void*)p;
        p += (bytes + 255) & ~(size_t)255;
        return r;
    };
    float* sums  = (float*)carve(128 * 4);
    float* stats = (float*)carve(128 * 4);
    float* act0  = (float*)carve((size_t)NB * 64 * 42 * 42 * 4);
    float* act1  = (float*)carve((size_t)NB * 64 * 21 * 21 * 4);
    float* act2  = (float*)carve((size_t)NB * 64 * 10 * 10 * 4);
    float* z     = (float*)carve((size_t)ND * 4);
    unsigned short* znbf = (unsigned short*)carve((size_t)ND * 2);
    float* sim   = (float*)carve((size_t)NB * NB * 4);
    int*   nbr   = (int*)carve((size_t)NB * 64 * 4);
    unsigned short* wt[12];
    for (int i = 0; i < 12; ++i) wt[i] = (unsigned short*)carve((size_t)DD * 2);
    unsigned short* zbf  = (unsigned short*)carve((size_t)ND * 2);
    float* qb    = (float*)carve((size_t)ND * 4);
    float* kb    = (float*)carve((size_t)ND * 4);
    float* vb    = (float*)carve((size_t)ND * 4);
    float* mb    = (float*)carve((size_t)ND * 4);
    unsigned short* mbf  = (unsigned short*)carve((size_t)ND * 2);
    float* mo    = (float*)carve((size_t)ND * 4);
    float* z1    = (float*)carve((size_t)ND * 4);
    unsigned short* z1bf = (unsigned short*)carve((size_t)ND * 2);
    float* h1    = (float*)carve((size_t)ND * 4);
    unsigned short* h1bf = (unsigned short*)carve((size_t)ND * 2);
    float* f2b   = (float*)carve((size_t)ND * 4);

    // ---- CNN backbone: 4 blocks, two-pass conv+BN, fused ReLU+pool ----
    struct Blk { const float* in; const float* w; int Cin, H, W; float* outp; };
    Blk blks[4] = {
        { x,    conv_w1,               3,  84, 84, act0 },
        { act0, conv_w + 0 * 64*64*9, 64,  42, 42, act1 },
        { act1, conv_w + 1 * 64*64*9, 64,  21, 21, act2 },
        { act2, conv_w + 2 * 64*64*9, 64,  10, 10, z    },
    };
    for (int i = 0; i < 4; ++i) {
        const Blk& B = blks[i];
        zero_f32<<<1, 128, 0, stream>>>(sums, 128);
        conv_stats<<<NB * 64, 256, 0, stream>>>(B.in, B.w, sums, B.Cin, B.H, B.W);
        finalize_stats<<<1, 64, 0, stream>>>(sums, stats,
            1.0f / ((float)NB * B.H * B.W));
        size_t total = (size_t)NB * 64 * (B.H >> 1) * (B.W >> 1);
        conv_bn_pool<<<(unsigned)((total + 255) / 256), 256, 0, stream>>>(
            B.in, B.w, stats, bn_g + i * 64, bn_b + i * 64, B.outp,
            NB, B.Cin, B.H, B.W);
    }
    // z = flattened conv4 features [256,1600] (NCHW flatten matches layout)

    // ---- kNN graph: zn = l2-normalize(z); sim = zn @ zn^T (WMMA); top-64 ----
    rownorm_bf16<<<NB, 256, 0, stream>>>(z, znbf, D);
    launch_gemm(znbf, znbf, sim, NB, NB, D, stream);
    topk64<<<NB, 256, 0, stream>>>(sim, nbr);

    // ---- Transpose+convert all GNN weights to bf16 [N,K] for GEMM_NT ----
    const float* wsrc[12] = {
        wq + 0*DD, wk + 0*DD, wv + 0*DD, wo + 0*DD, mlp_w1 + 0*DD, mlp_w2 + 0*DD,
        wq + 1*DD, wk + 1*DD, wv + 1*DD, wo + 1*DD, mlp_w1 + 1*DD, mlp_w2 + 1*DD,
    };
    for (int i = 0; i < 12; ++i)
        transpose_bf16<<<(DD + 255) / 256, 256, 0, stream>>>(wsrc[i], wt[i], D, D);

    // ---- GNN layers ----
    for (int l = 0; l < 2; ++l) {
        unsigned short* WqT = wt[l * 6 + 0];
        unsigned short* WkT = wt[l * 6 + 1];
        unsigned short* WvT = wt[l * 6 + 2];
        unsigned short* WoT = wt[l * 6 + 3];
        unsigned short* W1T = wt[l * 6 + 4];
        unsigned short* W2T = wt[l * 6 + 5];

        to_bf16<<<(ND + 255) / 256, 256, 0, stream>>>(z, zbf, ND);
        launch_gemm(zbf, WqT, qb, NB, D, D, stream);
        launch_gemm(zbf, WkT, kb, NB, D, D, stream);
        launch_gemm(zbf, WvT, vb, NB, D, D, stream);

        attn_fused<<<NB, 256, 0, stream>>>(qb, kb, vb, nbr, mb);

        to_bf16<<<(ND + 255) / 256, 256, 0, stream>>>(mb, mbf, ND);
        launch_gemm(mbf, WoT, mo, NB, D, D, stream);
        ln_residual<<<NB, 256, 0, stream>>>(z, mo, nullptr,
            ln1_g + l * D, ln1_b + l * D, z1, z1bf, D);

        launch_gemm(z1bf, W1T, h1, NB, D, D, stream);
        bias_relu_bf16<<<(ND + 255) / 256, 256, 0, stream>>>(
            h1, mlp_b1 + l * D, h1bf, D, ND);
        launch_gemm(h1bf, W2T, f2b, NB, D, D, stream);

        float* zdst = (l == 1) ? out : z;
        ln_residual<<<NB, 256, 0, stream>>>(z1, f2b, mlp_b2 + l * D,
            ln2_g + l * D, ln2_b + l * D, zdst, nullptr, D);
    }
}
